// GCN_10651518894757
// MI455X (gfx1250) — compile-verified, matrix operand order checked by reading, not run on hardware
//
#include <hip/hip_runtime.h>
#include <hip/hip_bf16.h>

#define HDIM 256

typedef __attribute__((ext_vector_type(16))) __bf16 v16bf;
typedef __attribute__((ext_vector_type(8)))  float  v8f;

// ---------- helpers ----------
__device__ __forceinline__ unsigned short f2bf(float f) {
    union { float f; unsigned u; } c; c.f = f;
    unsigned u = c.u;
    unsigned r = (u + 0x7FFFu + ((u >> 16) & 1u)) >> 16;   // round-to-nearest-even
    return (unsigned short)r;
}

__device__ __forceinline__ float bf2f_lo(unsigned u) {     // bits[15:0] -> f32
    union { unsigned u; float f; } c; c.u = u << 16; return c.f;
}
__device__ __forceinline__ float bf2f_hi(unsigned u) {     // bits[31:16] -> f32
    union { unsigned u; float f; } c; c.u = u & 0xFFFF0000u; return c.f;
}

// Load 16 bf16 K-values for one WMMA fragment lane (pure b128 loads, no VALU):
// rowp[kbase .. kbase+7] and rowp[kbase+16 .. kbase+23]
__device__ __forceinline__ v16bf load_frag_bf(const unsigned short* __restrict__ rowp, int kbase) {
    union { v16bf v; uint4 u[2]; } r;
    r.u[0] = *(const uint4*)(rowp + kbase);
    r.u[1] = *(const uint4*)(rowp + kbase + 16);
    return r.v;
}

// ---------- utility kernels ----------
__global__ void fill_zero_f4(float4* __restrict__ p, long long n4) {
    long long i = (long long)blockIdx.x * blockDim.x + threadIdx.x;
    long long stride = (long long)gridDim.x * blockDim.x;
    float4 z = make_float4(0.f, 0.f, 0.f, 0.f);
    for (; i < n4; i += stride) p[i] = z;
}

__global__ void cvt_idx(const long long* __restrict__ ei, int* __restrict__ si,
                        int* __restrict__ di, int E) {
    int e = blockIdx.x * blockDim.x + threadIdx.x;
    int stride = gridDim.x * blockDim.x;
    for (; e < E; e += stride) {
        si[e] = (int)ei[e];
        di[e] = (int)ei[E + e];
    }
}

__global__ void compute_deg(const int* __restrict__ di, float* __restrict__ deg, int E) {
    int e = blockIdx.x * blockDim.x + threadIdx.x;
    int stride = gridDim.x * blockDim.x;
    for (; e < E; e += stride) atomicAdd(deg + di[e], 1.0f);
}

// f32 -> bf16, 8 elements per thread
__global__ void cvt_bf16(const float* __restrict__ src, unsigned short* __restrict__ dst,
                         long long n8) {
    long long i = (long long)blockIdx.x * blockDim.x + threadIdx.x;
    long long stride = (long long)gridDim.x * blockDim.x;
    for (; i < n8; i += stride) {
        const float4 a = ((const float4*)src)[2 * i];
        const float4 b = ((const float4*)src)[2 * i + 1];
        uint4 o;
        o.x = (unsigned)f2bf(a.x) | ((unsigned)f2bf(a.y) << 16);
        o.y = (unsigned)f2bf(a.z) | ((unsigned)f2bf(a.w) << 16);
        o.z = (unsigned)f2bf(b.x) | ((unsigned)f2bf(b.y) << 16);
        o.w = (unsigned)f2bf(b.z) | ((unsigned)f2bf(b.w) << 16);
        ((uint4*)dst)[i] = o;
    }
}

// aggb[n][k] = bf16(agg[n][k] / max(deg[n],1)), 8 elements per thread
__global__ void norm_cvt(const float* __restrict__ agg, const float* __restrict__ deg,
                         unsigned short* __restrict__ aggb, int nnodes) {
    long long total = (long long)nnodes * (HDIM / 8);
    long long i = (long long)blockIdx.x * blockDim.x + threadIdx.x;
    long long stride = (long long)gridDim.x * blockDim.x;
    for (; i < total; i += stride) {
        int n = (int)(i >> 5);
        float rd = 1.0f / fmaxf(deg[n], 1.0f);
        const float4 a = ((const float4*)agg)[2 * i];
        const float4 b = ((const float4*)agg)[2 * i + 1];
        uint4 o;
        o.x = (unsigned)f2bf(a.x * rd) | ((unsigned)f2bf(a.y * rd) << 16);
        o.y = (unsigned)f2bf(a.z * rd) | ((unsigned)f2bf(a.w * rd) << 16);
        o.z = (unsigned)f2bf(b.x * rd) | ((unsigned)f2bf(b.y * rd) << 16);
        o.w = (unsigned)f2bf(b.z * rd) | ((unsigned)f2bf(b.w * rd) << 16);
        ((uint4*)aggb)[i] = o;
    }
}

// agg[dst] += float(hb[src]) : one 16B chunk (8 bf16) per thread, f32 atomics.
// 32 consecutive threads (one wave) cover one full edge row (512B load, 1KB atomics).
__global__ void scatter_add_bf16(const unsigned short* __restrict__ hb,
                                 const int* __restrict__ si, const int* __restrict__ di,
                                 float* __restrict__ agg, int E) {
    long long total = (long long)E * (HDIM / 8);
    long long i = (long long)blockIdx.x * blockDim.x + threadIdx.x;
    long long stride = (long long)gridDim.x * blockDim.x;
    for (; i < total; i += stride) {
        int e = (int)(i >> 5);
        int q = (int)(i & 31);
        int s = si[e];
        int d = di[e];
        uint4 v = ((const uint4*)(hb + (size_t)s * HDIM))[q];
        float* p = agg + (size_t)d * HDIM + q * 8;
        atomicAdd(p + 0, bf2f_lo(v.x));
        atomicAdd(p + 1, bf2f_hi(v.x));
        atomicAdd(p + 2, bf2f_lo(v.y));
        atomicAdd(p + 3, bf2f_hi(v.y));
        atomicAdd(p + 4, bf2f_lo(v.z));
        atomicAdd(p + 5, bf2f_hi(v.z));
        atomicAdd(p + 6, bf2f_lo(v.w));
        atomicAdd(p + 7, bf2f_hi(v.w));
    }
}

// ---------- fused SAGE layer GEMM (all-bf16 operands, f32 accumulate) ----------
// out = ELU( aggb @ Wlb^T + bl + hb @ Wrb^T )
// 256 threads = 8 waves: 4(M) x 2(N); wave tile = 16 rows x 128 cols (8 WMMA tiles).
// Per K-step: preload A + all 8 B fragments (one big load clause), then burst 8 WMMAs
// so in-order load returns pipeline against the WMMA issue (graduated s_wait_loadcnt).
__global__ void __launch_bounds__(256)
sage_gemm(const unsigned short* __restrict__ aggb, const unsigned short* __restrict__ hb,
          const unsigned short* __restrict__ Wlb, const unsigned short* __restrict__ Wrb,
          const float* __restrict__ bias, unsigned short* __restrict__ out_bf,
          float* __restrict__ out_f32, int nnodes) {
    const int lane = threadIdx.x & 31;
    const int w    = threadIdx.x >> 5;
    const int wm   = w >> 1;                 // 0..3
    const int wn   = w & 1;                  // 0..1
    const int mrow = blockIdx.x * 64 + wm * 16;
    const int ncol = wn * 128;
    const int hr   = lane & 15;              // row (A) / col (B,C) within tile
    const int koff = (lane < 16) ? 0 : 8;    // per-half-wave K chunk offset

    int arow = mrow + hr;
    if (arow >= nnodes) arow = nnodes - 1;   // clamp (store is guarded)

    const unsigned short* arow_p = aggb + (size_t)arow * HDIM;
    const unsigned short* hrow_p = hb + (size_t)arow * HDIM;
    const unsigned short* wlrow  = Wlb + (size_t)(ncol + hr) * HDIM;
    const unsigned short* wrrow  = Wrb + (size_t)(ncol + hr) * HDIM;

    v8f acc[8];
#pragma unroll
    for (int t = 0; t < 8; ++t) acc[t] = (v8f)(0.0f);

    // Phase 1: aggb @ Wlb^T   (B[k][n] = Wl[n][k] -> lane reads Wl row n contiguously)
#pragma unroll
    for (int k0 = 0; k0 < HDIM; k0 += 32) {
        const int kb = k0 + koff;
        v16bf a = load_frag_bf(arow_p, kb);
        v16bf b0 = load_frag_bf(wlrow + 0 * 16 * HDIM, kb);
        v16bf b1 = load_frag_bf(wlrow + 1 * 16 * HDIM, kb);
        v16bf b2 = load_frag_bf(wlrow + 2 * 16 * HDIM, kb);
        v16bf b3 = load_frag_bf(wlrow + 3 * 16 * HDIM, kb);
        v16bf b4 = load_frag_bf(wlrow + 4 * 16 * HDIM, kb);
        v16bf b5 = load_frag_bf(wlrow + 5 * 16 * HDIM, kb);
        v16bf b6 = load_frag_bf(wlrow + 6 * 16 * HDIM, kb);
        v16bf b7 = load_frag_bf(wlrow + 7 * 16 * HDIM, kb);
        acc[0] = __builtin_amdgcn_wmma_f32_16x16x32_bf16(false, a, false, b0, (short)0, acc[0], false, false);
        acc[1] = __builtin_amdgcn_wmma_f32_16x16x32_bf16(false, a, false, b1, (short)0, acc[1], false, false);
        acc[2] = __builtin_amdgcn_wmma_f32_16x16x32_bf16(false, a, false, b2, (short)0, acc[2], false, false);
        acc[3] = __builtin_amdgcn_wmma_f32_16x16x32_bf16(false, a, false, b3, (short)0, acc[3], false, false);
        acc[4] = __builtin_amdgcn_wmma_f32_16x16x32_bf16(false, a, false, b4, (short)0, acc[4], false, false);
        acc[5] = __builtin_amdgcn_wmma_f32_16x16x32_bf16(false, a, false, b5, (short)0, acc[5], false, false);
        acc[6] = __builtin_amdgcn_wmma_f32_16x16x32_bf16(false, a, false, b6, (short)0, acc[6], false, false);
        acc[7] = __builtin_amdgcn_wmma_f32_16x16x32_bf16(false, a, false, b7, (short)0, acc[7], false, false);
    }

    // Phase 2: hb @ Wrb^T
#pragma unroll
    for (int k0 = 0; k0 < HDIM; k0 += 32) {
        const int kb = k0 + koff;
        v16bf a = load_frag_bf(hrow_p, kb);
        v16bf b0 = load_frag_bf(wrrow + 0 * 16 * HDIM, kb);
        v16bf b1 = load_frag_bf(wrrow + 1 * 16 * HDIM, kb);
        v16bf b2 = load_frag_bf(wrrow + 2 * 16 * HDIM, kb);
        v16bf b3 = load_frag_bf(wrrow + 3 * 16 * HDIM, kb);
        v16bf b4 = load_frag_bf(wrrow + 4 * 16 * HDIM, kb);
        v16bf b5 = load_frag_bf(wrrow + 5 * 16 * HDIM, kb);
        v16bf b6 = load_frag_bf(wrrow + 6 * 16 * HDIM, kb);
        v16bf b7 = load_frag_bf(wrrow + 7 * 16 * HDIM, kb);
        acc[0] = __builtin_amdgcn_wmma_f32_16x16x32_bf16(false, a, false, b0, (short)0, acc[0], false, false);
        acc[1] = __builtin_amdgcn_wmma_f32_16x16x32_bf16(false, a, false, b1, (short)0, acc[1], false, false);
        acc[2] = __builtin_amdgcn_wmma_f32_16x16x32_bf16(false, a, false, b2, (short)0, acc[2], false, false);
        acc[3] = __builtin_amdgcn_wmma_f32_16x16x32_bf16(false, a, false, b3, (short)0, acc[3], false, false);
        acc[4] = __builtin_amdgcn_wmma_f32_16x16x32_bf16(false, a, false, b4, (short)0, acc[4], false, false);
        acc[5] = __builtin_amdgcn_wmma_f32_16x16x32_bf16(false, a, false, b5, (short)0, acc[5], false, false);
        acc[6] = __builtin_amdgcn_wmma_f32_16x16x32_bf16(false, a, false, b6, (short)0, acc[6], false, false);
        acc[7] = __builtin_amdgcn_wmma_f32_16x16x32_bf16(false, a, false, b7, (short)0, acc[7], false, false);
    }

    // Epilogue: bias + ELU + store (C layout: lane<16 -> M=v, lane>=16 -> M=v+8; N=lane&15)
    const int rbase = mrow + ((lane < 16) ? 0 : 8);
#pragma unroll
    for (int t = 0; t < 8; ++t) {
        const int n = ncol + t * 16 + hr;
        const float b = bias[n];
#pragma unroll
        for (int v = 0; v < 8; ++v) {
            const int r = rbase + v;
            if (r < nnodes) {
                float x = acc[t][v] + b;
                float y = (x > 0.0f) ? x : (__expf(x) - 1.0f);
                if (out_bf)  out_bf[(size_t)r * HDIM + n] = f2bf(y);
                if (out_f32) out_f32[(size_t)r * HDIM + n] = y;
            }
        }
    }
}

// ---------- launch ----------
extern "C" void kernel_launch(void* const* d_in, const int* in_sizes, int n_in,
                              void* d_out, int out_size, void* d_ws, size_t ws_size,
                              hipStream_t stream) {
    const float*     x  = (const float*)d_in[0];
    const long long* ei = (const long long*)d_in[1];
    const float* Wl[3] = {(const float*)d_in[2], (const float*)d_in[4], (const float*)d_in[6]};
    const float* Wr[3] = {(const float*)d_in[3], (const float*)d_in[5], (const float*)d_in[7]};
    const float* bl[3] = {(const float*)d_in[8], (const float*)d_in[9], (const float*)d_in[10]};
    float* out = (float*)d_out;

    const int N = in_sizes[0] / HDIM;    // 50000
    const int E = in_sizes[1] / 2;       // 800000
    const long long NH = (long long)N * HDIM;
    const long long HH = (long long)HDIM * HDIM;

    // ---- workspace layout ----
    float* agg = (float*)d_ws;                        // NH f32
    float* deg = agg + NH;                            // N  f32
    int*   si  = (int*)(deg + N);                     // E  i32
    int*   di  = si + E;                              // E  i32
    unsigned short* hb0  = (unsigned short*)(di + E); // NH bf16
    unsigned short* hb1  = hb0 + NH;                  // NH bf16
    unsigned short* aggb = hb1 + NH;                  // NH bf16
    unsigned short* wb   = aggb + NH;                 // 6*H*H bf16

    const int T = 256;
    const int gemm_blocks = (N + 63) / 64;

    // ---- one-time per call: indices, degree, bf16 weights, bf16 input ----
    cvt_idx<<<(E + T - 1) / T, T, 0, stream>>>(ei, si, di, E);
    fill_zero_f4<<<64, T, 0, stream>>>((float4*)deg, N / 4);
    compute_deg<<<(E + T - 1) / T, T, 0, stream>>>(di, deg, E);
    for (int l = 0; l < 3; ++l) {
        cvt_bf16<<<64, T, 0, stream>>>(Wl[l], wb + (2 * l) * HH, HH / 8);
        cvt_bf16<<<64, T, 0, stream>>>(Wr[l], wb + (2 * l + 1) * HH, HH / 8);
    }
    cvt_bf16<<<2048, T, 0, stream>>>(x, hb0, NH / 8);

    // ---- layers: hb0 -> hb1 -> hb0(reused) -> d_out(f32) ----
    const unsigned short* hin[3] = {hb0, hb1, hb0};
    unsigned short*      hout[3] = {hb1, hb0, nullptr};
    float*               fout[3] = {nullptr, nullptr, out};

    for (int l = 0; l < 3; ++l) {
        fill_zero_f4<<<4096, T, 0, stream>>>((float4*)agg, NH / 4);
        scatter_add_bf16<<<8192, T, 0, stream>>>(hin[l], si, di, agg, E);
        norm_cvt<<<2048, T, 0, stream>>>(agg, deg, aggb, N);
        sage_gemm<<<gemm_blocks, T, 0, stream>>>(aggb, hin[l], wb + (2 * l) * HH,
                                                 wb + (2 * l + 1) * HH, bl[l],
                                                 hout[l], fout[l], N);
    }
}
